// Model_12652973654024
// MI455X (gfx1250) — matrix operand order repaired and checked by reading, not compiled, not run on hardware
//
#include <hip/hip_runtime.h>

#define SEQ    1024
#define DMODEL 768
#define NHEAD  12
#define NLAYER 12

typedef __attribute__((ext_vector_type(16))) __bf16 v16bf;
typedef __attribute__((ext_vector_type(8)))  float  v8f;
typedef __attribute__((ext_vector_type(4)))  unsigned int u32x4;
typedef __attribute__((ext_vector_type(8)))  int    i32x8;
typedef __attribute__((ext_vector_type(4)))  int    i32x4;

#if defined(__HIP_DEVICE_COMPILE__) && __has_builtin(__builtin_amdgcn_tensor_load_to_lds) && __has_builtin(__builtin_amdgcn_s_wait_tensorcnt)
#define HAS_TDM 1
#else
#define HAS_TDM 0
#endif

__device__ __forceinline__ unsigned short f2bf(float f) {
  union { float f; unsigned int u; } v; v.f = f;
  unsigned int r = v.u + 0x7FFFu + ((v.u >> 16) & 1u);   // round-to-nearest-even
  return (unsigned short)(r >> 16);
}
__device__ __forceinline__ unsigned int pack2bf(float a, float b) {
  return (unsigned int)f2bf(a) | ((unsigned int)f2bf(b) << 16);
}
__device__ __forceinline__ float gelu_tanh(float x) {
  float t = tanhf(0.7978845608028654f * (x + 0.044715f * x * x * x));
  return 0.5f * x * (1.0f + t);
}
__device__ __forceinline__ v8f zero_v8f() {
  v8f z = {0.f, 0.f, 0.f, 0.f, 0.f, 0.f, 0.f, 0.f};
  return z;
}

// ---------------------------------------------------------------------------
// TDM: issue a 2D tile copy (bf16 elements) global -> LDS via the Tensor Data
// Mover.  D# packed per CDNA5 ISA 8.3/8.4 (data_size=1 -> 2 bytes, type=2).
// ---------------------------------------------------------------------------
__device__ __forceinline__ void tdm_load_tile_2d(unsigned lds_addr, const void* g,
                                                 int tdim0, int tdim1,
                                                 long stride0, int tile0, int tile1) {
#if HAS_TDM
  unsigned long long ga = (unsigned long long)g;
  u32x4 g0;
  g0[0] = 1u;                                             // count=1, user mode
  g0[1] = lds_addr;                                       // lds_addr (bytes)
  g0[2] = (unsigned int)ga;                               // global_addr[31:0]
  g0[3] = (unsigned int)((ga >> 32) & 0x01FFFFFFu) | (2u << 30); // ga[56:32] | type=2
  i32x8 g1;
  g1[0] = 0x00010000;                                     // data_size=1 (2B), mask=0
  g1[1] = (int)(((unsigned)tdim0 & 0xFFFFu) << 16);       // tensor_dim0[15:0] @63:48
  g1[2] = (int)((((unsigned)tdim0 >> 16) & 0xFFFFu) |     // tensor_dim0[31:16]
                (((unsigned)tdim1 & 0xFFFFu) << 16));     // tensor_dim1[15:0]
  g1[3] = (int)((((unsigned)tdim1 >> 16) & 0xFFFFu) |     // tensor_dim1[31:16]
                (((unsigned)tile0 & 0xFFFFu) << 16));     // tile_dim0
  g1[4] = (int)((unsigned)tile1 & 0xFFFFu);               // tile_dim1 (tile_dim2=0)
  g1[5] = (int)(unsigned)(stride0 & 0xFFFFFFFFl);         // tensor_dim0_stride[31:0]
  g1[6] = (int)((stride0 >> 32) & 0xFFFFl);               // tensor_dim0_stride[47:32]
  g1[7] = 0;
  i32x4 z4 = {0, 0, 0, 0};
#if __clang_major__ >= 23
  i32x8 z8 = {0, 0, 0, 0, 0, 0, 0, 0};
  __builtin_amdgcn_tensor_load_to_lds(g0, g1, z4, z4, z8, 0);
#else
  __builtin_amdgcn_tensor_load_to_lds(g0, g1, z4, z4, 0);
#endif
#else
  (void)lds_addr; (void)g; (void)tdim0; (void)tdim1; (void)stride0; (void)tile0; (void)tile1;
#endif
}

// ---------------------------------------------------------------------------
// Embedding: x = wte[ids] + wpe ; also writes layers_output[0]
// ---------------------------------------------------------------------------
__global__ __launch_bounds__(256)
void embed_kernel(const int* __restrict__ ids, const float* __restrict__ wte,
                  const float* __restrict__ wpe, float* __restrict__ x,
                  float* __restrict__ l0) {
  const int s = blockIdx.x;
  const long id = (long)ids[s];
  for (int d = threadIdx.x; d < DMODEL; d += 256) {
    float v = wte[id * DMODEL + d] + wpe[(size_t)s * DMODEL + d];
    x[(size_t)s * DMODEL + d]  = v;
    l0[(size_t)s * DMODEL + d] = v;
  }
}

// ---------------------------------------------------------------------------
// LayerNorm: one block per row.  BF16OUT=1 writes bf16 (GEMM A-operand path).
// ---------------------------------------------------------------------------
template <int BF16OUT>
__global__ __launch_bounds__(256)
void ln_kernel(const float* __restrict__ x, const float* __restrict__ w,
               const float* __restrict__ b, void* __restrict__ yv) {
  __shared__ float red[256];
  const int s = blockIdx.x, tid = threadIdx.x;
  const float* xr = x + (size_t)s * DMODEL;

  float p = 0.f;
  for (int d = tid; d < DMODEL; d += 256) p += xr[d];
  red[tid] = p; __syncthreads();
  for (int st = 128; st > 0; st >>= 1) { if (tid < st) red[tid] += red[tid + st]; __syncthreads(); }
  const float mean = red[0] * (1.0f / DMODEL);
  __syncthreads();

  float v = 0.f;
  for (int d = tid; d < DMODEL; d += 256) { float dd = xr[d] - mean; v += dd * dd; }
  red[tid] = v; __syncthreads();
  for (int st = 128; st > 0; st >>= 1) { if (tid < st) red[tid] += red[tid + st]; __syncthreads(); }
  const float rstd = rsqrtf(red[0] * (1.0f / DMODEL) + 1e-5f);

  for (int d = tid; d < DMODEL; d += 256) {
    float o = (xr[d] - mean) * rstd * w[d] + b[d];
    if (BF16OUT) ((unsigned short*)yv)[(size_t)s * DMODEL + d] = f2bf(o);
    else         ((float*)yv)[(size_t)s * DMODEL + d] = o;
  }
}

// ---------------------------------------------------------------------------
// Weight pack: W[K,N] fp32 -> Wt[N,K] bf16 (32x32 LDS-tiled transpose)
// ---------------------------------------------------------------------------
__global__ __launch_bounds__(256)
void wpack_kernel(const float* __restrict__ W, unsigned short* __restrict__ Wt,
                  int K, int N) {
  __shared__ unsigned short tile[32][33];
  const int tx = threadIdx.x & 31, ty = threadIdx.x >> 5;  // 32 x 8
  const int n0 = blockIdx.x * 32, k0 = blockIdx.y * 32;
  for (int r = ty; r < 32; r += 8)
    tile[r][tx] = f2bf(W[(size_t)(k0 + r) * N + n0 + tx]);
  __syncthreads();
  for (int r = ty; r < 32; r += 8)
    Wt[(size_t)(n0 + r) * K + k0 + tx] = tile[tx][r];
}

// ---------------------------------------------------------------------------
// bf16 WMMA GEMM: C = A[M,K](bf16) @ Bt[N,K](bf16)^T + bias
// 128x128 block tile, BK=32, 8 waves: each wave 2x4 WMMA 16x16 tiles.
// Double-buffered LDS staging via the Tensor Data Mover: wave 0 issues the
// next k-slab's 2 descriptors, then s_wait_tensorcnt(2) — in-order TENSORcnt
// completion guarantees the current slab is resident while the next streams.
// MODE: 1 = fp32 out, +residual   2 = bf16 out, gelu   3 = bf16 qkv split
// ---------------------------------------------------------------------------
template <int MODE>
__global__ __launch_bounds__(256)
void gemm_bf16(const unsigned short* __restrict__ A, const unsigned short* __restrict__ Bt,
               const float* __restrict__ bias, const float* R,
               void* Cv, int N, int K) {
  __shared__ unsigned short sA[2][128 * 32];   // [row][k] bf16, ping-pong
  __shared__ unsigned short sBt[2][128 * 32];  // [col][k] bf16, ping-pong

  const int tid  = threadIdx.x;
  const int lane = tid & 31;
  const int w    = tid >> 5;
  const int wr   = w >> 1;                       // 0..3 -> 32-row strip
  const int wc   = w & 1;                        // 0..1 -> 64-col strip
  const int ml   = lane & 15;
  const int kbA  = (lane & 16) ? 8 : 0;          // A half-wave K base
  const int kbB  = (lane & 16) ? 8 : 0;          // B half-wave K base
  const int m0   = blockIdx.y * 128;
  const int n0   = blockIdx.x * 128;
  const int kofsA[8] = {0, 2, 4, 6, 16, 18, 20, 22};

  v8f acc[2][4];
#pragma unroll
  for (int mt = 0; mt < 2; ++mt)
#pragma unroll
    for (int nt = 0; nt < 4; ++nt) acc[mt][nt] = zero_v8f();

  // stage one k-slab (32 columns of A, 32 columns of Bt) into buffer `buf`
  auto stage = [&](int buf, int k0) {
#if HAS_TDM
    if (tid < 32) {   // wave 0 drives the Tensor Data Mover for both tiles
      tdm_load_tile_2d((unsigned)(unsigned long long)(void*)&sA[buf][0],
                       A + (size_t)m0 * K + k0, K, SEQ, (long)K, 32, 128);
      tdm_load_tile_2d((unsigned)(unsigned long long)(void*)&sBt[buf][0],
                       Bt + (size_t)n0 * K + k0, K, N, (long)K, 32, 128);
    }
#else
#pragma unroll
    for (int j = 0; j < 8; ++j) {          // 2048 dwords, 8 per thread
      int i = tid + 256 * j;
      int r = i >> 4, c2 = (i & 15) * 2;
      *(unsigned int*)(&sA[buf][0] + r * 32 + c2)  = *(const unsigned int*)(A  + (size_t)(m0 + r) * K + k0 + c2);
      *(unsigned int*)(&sBt[buf][0] + r * 32 + c2) = *(const unsigned int*)(Bt + (size_t)(n0 + r) * K + k0 + c2);
    }
#endif
  };

  // one k-slab of WMMA work from buffer `buf`
  auto compute = [&](int buf) {
    union { v16bf v; unsigned int u[8]; } af[2], bfr[4];
#pragma unroll
    for (int mt = 0; mt < 2; ++mt) {
      int m = wr * 32 + mt * 16 + ml;
#pragma unroll
      for (int j = 0; j < 8; ++j)
        af[mt].u[j] = *(const unsigned int*)(&sA[buf][0] + m * 32 + kbA + kofsA[j]);
    }
#pragma unroll
    for (int nt = 0; nt < 4; ++nt) {
      int n = wc * 64 + nt * 16 + ml;
#pragma unroll
      for (int j = 0; j < 8; ++j)
        bfr[nt].u[j] = *(const unsigned int*)(&sBt[buf][0] + n * 32 + kbB + kofsA[j]);
    }
#pragma unroll
    for (int mt = 0; mt < 2; ++mt)
#pragma unroll
      for (int nt = 0; nt < 4; ++nt)
        acc[mt][nt] = __builtin_amdgcn_wmma_f32_16x16x32_bf16(
            false, af[mt].v, false, bfr[nt].v, (short)0, acc[mt][nt], false, false);
  };

  // software pipeline: prologue + steady state + peeled last slab
  stage(0, 0);
  int cur = 0;
  for (int k0 = 0; k0 < K - 32; k0 += 32) {
    stage(cur ^ 1, k0 + 32);        // overlap next slab's DMA with this WMMA
#if HAS_TDM
    if (tid < 32) __builtin_amdgcn_s_wait_tensorcnt(2);  // current slab resident
#endif
    __syncthreads();
    compute(cur);
    __syncthreads();
    cur ^= 1;
  }
#if HAS_TDM
  if (tid < 32) __builtin_amdgcn_s_wait_tensorcnt(0);
#endif
  __syncthreads();
  compute(cur);

  const int mrow = (lane & 16) ? 8 : 0;
#pragma unroll
  for (int mt = 0; mt < 2; ++mt) {
#pragma unroll
    for (int nt = 0; nt < 4; ++nt) {
      int gn = n0 + wc * 64 + nt * 16 + ml;
      float bv = bias[gn];
#pragma unroll
      for (int i = 0; i < 8; ++i) {
        int gm = m0 + wr * 32 + mt * 16 + mrow + i;
        float v = acc[mt][nt][i] + bv;
        if (MODE == 1) {
          ((float*)Cv)[(size_t)gm * N + gn] = v + R[(size_t)gm * N + gn];
        } else if (MODE == 2) {
          ((unsigned short*)Cv)[(size_t)gm * N + gn] = f2bf(gelu_tanh(v));
        } else {  // MODE 3: split QKV columns into q/k/v bf16 planes
          int seg  = gn / DMODEL;
          int dcol = gn - seg * DMODEL;
          ((unsigned short*)Cv)[(size_t)seg * (SEQ * DMODEL) + (size_t)gm * DMODEL + dcol] = f2bf(v);
        }
      }
    }
  }
}

// ---------------------------------------------------------------------------
// Causal attention: one wave per (16-row tile, head).
// scores (16x1024 f32 = 64KB LDS) via WMMA, row softmax, P@V via WMMA.
// Output written bf16 (it is the next GEMM's A operand).
// ---------------------------------------------------------------------------
__global__ __launch_bounds__(32)
void attn_kernel(const unsigned short* __restrict__ qb,
                 const unsigned short* __restrict__ kb,
                 const unsigned short* __restrict__ vb,
                 unsigned short* __restrict__ ob) {
  __shared__ float sc[16 * 1024];
  const int lane = threadIdx.x;
  const int rt   = blockIdx.x;     // 0..63 row tile
  const int h    = blockIdx.y;     // head
  const int ml   = lane & 15;
  const int kbA  = (lane & 16) ? 8 : 0;
  const int kbB  = (lane & 16) ? 8 : 0;
  const int mrow = (lane & 16) ? 8 : 0;
  const int kofsA[8] = {0, 2, 4, 6, 16, 18, 20, 22};

  // ---- scores = (Q K^T)/8 with causal mask ----
  const int qrow = rt * 16 + ml;
  for (int tt = 0; tt < 64; ++tt) {
    v8f acc = zero_v8f();
#pragma unroll
    for (int ds = 0; ds < 2; ++ds) {
      union { v16bf v; unsigned int u[8]; } a, b;
      const int tcol = tt * 16 + ml;
#pragma unroll
      for (int j = 0; j < 8; ++j) {
        a.u[j] = *(const unsigned int*)(qb + (size_t)qrow * DMODEL + h * 64 + ds * 32 + kbA + kofsA[j]);
        b.u[j] = *(const unsigned int*)(kb + (size_t)tcol * DMODEL + h * 64 + ds * 32 + kbB + kofsA[j]);
      }
      acc = __builtin_amdgcn_wmma_f32_16x16x32_bf16(false, a.v, false, b.v, (short)0, acc, false, false);
    }
#pragma unroll
    for (int i = 0; i < 8; ++i) {
      int r = rt * 16 + mrow + i;
      int t = tt * 16 + ml;
      float v = acc[i] * 0.125f;
      if (t > r) v = -3.0e38f;
      sc[(mrow + i) * 1024 + t] = v;
    }
  }

  // ---- per-row softmax; lane keeps 1/sum of its fragment row (ml) ----
  float myinv = 0.f;
  for (int m = 0; m < 16; ++m) {
    float mx = -3.4e38f;
    for (int t = lane; t < 1024; t += 32) mx = fmaxf(mx, sc[m * 1024 + t]);
#pragma unroll
    for (int s = 16; s > 0; s >>= 1) mx = fmaxf(mx, __shfl_xor(mx, s, 32));
    float sum = 0.f;
    for (int t = lane; t < 1024; t += 32) {
      float e = __expf(sc[m * 1024 + t] - mx);
      sc[m * 1024 + t] = e;
      sum += e;
    }
#pragma unroll
    for (int s = 16; s > 0; s >>= 1) sum += __shfl_xor(sum, s, 32);
    if (ml == m) myinv = 1.0f / sum;
  }

  // ---- O = P @ V ----
  v8f oacc[4];
#pragma unroll
  for (int nt = 0; nt < 4; ++nt) oacc[nt] = zero_v8f();
  for (int ks = 0; ks < 32; ++ks) {
    union { v16bf v; unsigned int u[8]; } p;
#pragma unroll
    for (int j = 0; j < 8; ++j) {
      int kk = ks * 32 + kbA + kofsA[j];
      p.u[j] = pack2bf(sc[ml * 1024 + kk] * myinv, sc[ml * 1024 + kk + 1] * myinv);
    }
#pragma unroll
    for (int nt = 0; nt < 4; ++nt) {
      union { v16bf v; unsigned int u[8]; } bv;
      const int col = h * 64 + nt * 16 + ml;
#pragma unroll
      for (int j = 0; j < 8; ++j) {
        int t0 = ks * 32 + kbB + kofsA[j];
        bv.u[j] = (unsigned int)vb[(size_t)t0 * DMODEL + col] |
                  ((unsigned int)vb[(size_t)(t0 + 1) * DMODEL + col] << 16);
      }
      oacc[nt] = __builtin_amdgcn_wmma_f32_16x16x32_bf16(false, p.v, false, bv.v, (short)0, oacc[nt], false, false);
    }
  }
#pragma unroll
  for (int nt = 0; nt < 4; ++nt) {
    const int col = h * 64 + nt * 16 + ml;
#pragma unroll
    for (int i = 0; i < 8; ++i) {
      int r = rt * 16 + mrow + i;
      ob[(size_t)r * DMODEL + col] = f2bf(oacc[nt][i]);
    }
  }
}

// ---------------------------------------------------------------------------
extern "C" void kernel_launch(void* const* d_in, const int* in_sizes, int n_in,
                              void* d_out, int out_size, void* d_ws, size_t ws_size,
                              hipStream_t stream) {
  (void)in_sizes; (void)n_in; (void)out_size; (void)ws_size;
  const int S = SEQ, Dm = DMODEL;
  const size_t SD = (size_t)S * Dm;

  const int*   ids   = (const int*)d_in[0];
  const float* wte   = (const float*)d_in[1];
  const float* wpe   = (const float*)d_in[2];
  const float* ln1w  = (const float*)d_in[3];
  const float* ln1b  = (const float*)d_in[4];
  const float* qkvw  = (const float*)d_in[5];
  const float* qkvb  = (const float*)d_in[6];
  const float* outw  = (const float*)d_in[7];
  const float* outb  = (const float*)d_in[8];
  const float* ln2w  = (const float*)d_in[9];
  const float* ln2b  = (const float*)d_in[10];
  const float* upw   = (const float*)d_in[11];
  const float* upb   = (const float*)d_in[12];
  const float* downw = (const float*)d_in[13];
  const float* downb = (const float*)d_in[14];
  const float* lnfw  = (const float*)d_in[15];
  const float* lnfb  = (const float*)d_in[16];

  float* x = (float*)d_ws;                              // fp32 residual stream
  unsigned short* xnb     = (unsigned short*)(x + SD);  // bf16 LN output
  unsigned short* attnb   = xnb + SD;                   // bf16 attention output
  unsigned short* hiddenb = attnb + SD;                 // bf16 gelu(up) output
  unsigned short* qkvbuf  = hiddenb + 4 * SD;           // q | k | v bf16 planes
  unsigned short* qkvwT   = qkvbuf + 3 * SD;            // bf16 [N,K] packed weights
  unsigned short* outwT   = qkvwT + (size_t)Dm * 3 * Dm;
  unsigned short* upwT    = outwT + (size_t)Dm * Dm;
  unsigned short* downwT  = upwT + (size_t)Dm * 4 * Dm;

  float* final_ln = (float*)d_out;
  float* layers   = final_ln + SD;   // [13, S, D]

  embed_kernel<<<S, 256, 0, stream>>>(ids, wte, wpe, x, layers);

  for (int l = 0; l < NLAYER; ++l) {
    // pack this layer's weights: fp32 [K,N] -> bf16 [N,K]
    wpack_kernel<<<dim3(3 * Dm / 32, Dm / 32), 256, 0, stream>>>(qkvw + (size_t)l * Dm * 3 * Dm, qkvwT, Dm, 3 * Dm);
    wpack_kernel<<<dim3(Dm / 32, Dm / 32), 256, 0, stream>>>(outw + (size_t)l * Dm * Dm, outwT, Dm, Dm);
    wpack_kernel<<<dim3(4 * Dm / 32, Dm / 32), 256, 0, stream>>>(upw + (size_t)l * Dm * 4 * Dm, upwT, Dm, 4 * Dm);
    wpack_kernel<<<dim3(Dm / 32, 4 * Dm / 32), 256, 0, stream>>>(downw + (size_t)l * 4 * Dm * Dm, downwT, 4 * Dm, Dm);

    ln_kernel<1><<<S, 256, 0, stream>>>(x, ln1w + (size_t)l * Dm, ln1b + (size_t)l * Dm, xnb);
    gemm_bf16<3><<<dim3(18, 8), 256, 0, stream>>>(
        xnb, qkvwT, qkvb + (size_t)l * 3 * Dm, nullptr, qkvbuf, 3 * Dm, Dm);
    attn_kernel<<<dim3(64, NHEAD), 32, 0, stream>>>(qkvbuf, qkvbuf + SD, qkvbuf + 2 * SD, attnb);
    gemm_bf16<1><<<dim3(6, 8), 256, 0, stream>>>(
        attnb, outwT, outb + (size_t)l * Dm, x, x, Dm, Dm);
    ln_kernel<1><<<S, 256, 0, stream>>>(x, ln2w + (size_t)l * Dm, ln2b + (size_t)l * Dm, xnb);
    gemm_bf16<2><<<dim3(24, 8), 256, 0, stream>>>(
        xnb, upwT, upb + (size_t)l * 4 * Dm, nullptr, hiddenb, 4 * Dm, Dm);
    gemm_bf16<1><<<dim3(6, 8), 256, 0, stream>>>(
        hiddenb, downwT, downb + (size_t)l * Dm, x, x, Dm, 4 * Dm);
    hipMemcpyAsync(layers + (size_t)(l + 1) * SD, x, SD * sizeof(float),
                   hipMemcpyDeviceToDevice, stream);
  }
  ln_kernel<0><<<S, 256, 0, stream>>>(x, lnfw, lnfb, final_ln);
}
